// PointerNet_64725157150945
// MI455X (gfx1250) — matrix operand order repaired and checked by reading, not compiled, not run on hardware
//
#include <hip/hip_runtime.h>
#include <stdint.h>
#include <stddef.h>

// ---------------- types ----------------
typedef __bf16 bf16;
typedef __attribute__((ext_vector_type(16))) __bf16 v16bf;
typedef __attribute__((ext_vector_type(8)))  __bf16 v8bf;
typedef __attribute__((ext_vector_type(8)))  float  v8f;

// ---------------- problem constants ----------------
#define S_   128
#define B_   128
#define NS_  16
#define D_   256
#define H_   256
#define L_   2
#define BN_  (B_ * NS_)     // 2048
#define H4_  (4 * H_)       // 1024
#define CLIP_ 10.0f
#define NEGINF_ (-1.0e30f)

// =====================================================================
// WMMA fragment loaders (ISA 7.12.2, bf16, wave32)
//  A 16x32 (MxK): lane L holds row M=L&15, K in [khb,khb+8) and [16+khb,16+khb+8), khb=(L/16)*8
//  B 32x16 (KxN), weights stored row-major as W[N,K]: lane L holds col N=L&15, K in [kb,kb+8) and [16+kb,16+kb+8), kb=(L/16)*8
// =====================================================================
__device__ inline v16bf load_a_frag(const bf16* __restrict__ A, int lda, int m0, int k0, int lane) {
    int m   = m0 + (lane & 15);
    int khb = (lane >> 4) * 8;
    const bf16* pa = A + (size_t)m * lda + k0;
    v8bf c0 = *(const v8bf*)(pa + khb);
    v8bf c1 = *(const v8bf*)(pa + 16 + khb);
    v16bf a;
#pragma unroll
    for (int i = 0; i < 8; ++i) { a[i] = c0[i]; a[i + 8] = c1[i]; }
    return a;
}

__device__ inline v16bf load_b_frag(const bf16* __restrict__ Wt, int ldw, int n0, int k0, int lane) {
    int n  = n0 + (lane & 15);
    int kb = k0 + (lane >> 4) * 8;
    const bf16* pb = Wt + (size_t)n * ldw + kb;
    v8bf c0 = *(const v8bf*)(pb);
    v8bf c1 = *(const v8bf*)(pb + 16);
    v16bf b;
#pragma unroll
    for (int i = 0; i < 8; ++i) { b[i] = c0[i]; b[i + 8] = c1[i]; }
    return b;
}

// =====================================================================
// GEMM:  C[M,N] = A1[M,K]*W1[N,K]^T (+ A2*W2^T) (+ bias[N])
// bf16 in, f32 accumulate via v_wmma_f32_16x16x32_bf16.
// Template: NT = 16-wide N-tiles per wave (1 or 4; A-frag reused NT times),
//           DUAL = second A/W product, OUTBF16 = bf16 vs f32 output.
// grid = (N/(64*NT), M/16), block = 128 (4 waves);
// wave w owns tile m0=by*16, n0=(bx*4+w)*16*NT .. +16*NT
// =====================================================================
template <int NT, bool DUAL, bool OUTBF16>
__global__ void gemm_wmma(const bf16* __restrict__ A1, const bf16* __restrict__ W1,
                          const bf16* __restrict__ A2, const bf16* __restrict__ W2,
                          const float* __restrict__ bias,
                          void* __restrict__ Cout,
                          int M, int N, int K) {
    int lane = threadIdx.x & 31;
    int wave = threadIdx.x >> 5;
    int n0 = (blockIdx.x * 4 + wave) * (16 * NT);
    int m0 = blockIdx.y * 16;
    if (n0 >= N || m0 >= M) return;

    v8f acc[NT];
#pragma unroll
    for (int j = 0; j < NT; ++j) acc[j] = (v8f){};

#pragma unroll 2
    for (int k0 = 0; k0 < K; k0 += 32) {
        v16bf a1 = load_a_frag(A1, K, m0, k0, lane);
#pragma unroll
        for (int j = 0; j < NT; ++j) {
            v16bf b1 = load_b_frag(W1, K, n0 + 16 * j, k0, lane);
            acc[j] = __builtin_amdgcn_wmma_f32_16x16x32_bf16(false, a1, false, b1,
                                                             (short)0, acc[j], false, false);
        }
        if (DUAL) {
            v16bf a2 = load_a_frag(A2, K, m0, k0, lane);
#pragma unroll
            for (int j = 0; j < NT; ++j) {
                v16bf b2 = load_b_frag(W2, K, n0 + 16 * j, k0, lane);
                acc[j] = __builtin_amdgcn_wmma_f32_16x16x32_bf16(false, a2, false, b2,
                                                                 (short)0, acc[j], false, false);
            }
        }
    }

#pragma unroll
    for (int j = 0; j < NT; ++j) {
        int n = n0 + 16 * j + (lane & 15);
        float bv = bias ? bias[n] : 0.0f;
#pragma unroll
        for (int r = 0; r < 8; ++r) {
            int m = m0 + r + 8 * (lane >> 4);
            float v = acc[j][r] + bv;
            if (OUTBF16) ((bf16*)Cout)[(size_t)m * N + n] = (bf16)v;
            else         ((float*)Cout)[(size_t)m * N + n] = v;
        }
    }
}

// =====================================================================
// LSTM cell elementwise: z[M,4H] (gate order i,f,g,o), c f32 in/out, h -> bf16
// =====================================================================
__global__ void lstm_cell_kernel(const float* __restrict__ z, float* __restrict__ c,
                                 bf16* __restrict__ h, bf16* __restrict__ hcopy, int MH) {
    int i = blockIdx.x * blockDim.x + threadIdx.x;
    if (i >= MH) return;
    int m  = i / H_;
    int hh = i % H_;
    const float* zr = z + (size_t)m * H4_;
    float zi = zr[hh], zf = zr[H_ + hh], zg = zr[2 * H_ + hh], zo = zr[3 * H_ + hh];
    float si = 1.0f / (1.0f + __expf(-zi));
    float sf = 1.0f / (1.0f + __expf(-zf));
    float so = 1.0f / (1.0f + __expf(-zo));
    float cn = sf * c[i] + si * tanhf(zg);
    c[i] = cn;
    float hn = so * tanhf(cn);
    h[i] = (bf16)hn;
    if (hcopy) hcopy[i] = (bf16)hn;
}

// ---------------- small elementwise helpers ----------------
__global__ void f32_to_bf16_kernel(const float* __restrict__ x, bf16* __restrict__ y, int n) {
    int i = blockIdx.x * blockDim.x + threadIdx.x;
    if (i < n) y[i] = (bf16)x[i];
}
__global__ void zero_bf16_kernel(bf16* __restrict__ x, int n) {
    int i = blockIdx.x * blockDim.x + threadIdx.x;
    if (i < n) x[i] = (bf16)0.0f;
}
__global__ void zero_f32_kernel(float* __restrict__ x, int n) {
    int i = blockIdx.x * blockDim.x + threadIdx.x;
    if (i < n) x[i] = 0.0f;
}
// repeat encoder final state over NS samples: [L,B,H] -> [L,Bn,H]
__global__ void init_dec_state_kernel(const bf16* __restrict__ h_enc, const float* __restrict__ c_enc,
                                      bf16* __restrict__ h_dec, float* __restrict__ c_dec) {
    int i = blockIdx.x * blockDim.x + threadIdx.x;
    if (i >= L_ * BN_ * H_) return;
    int l  = i / (BN_ * H_);
    int r  = i % (BN_ * H_);
    int bn = r / H_;
    int hh = r % H_;
    int b  = bn / NS_;
    size_t src = ((size_t)l * B_ + b) * H_ + hh;
    h_dec[i] = h_enc[src];
    c_dec[i] = c_enc[src];
}
__global__ void init_decx_kernel(const float* __restrict__ init_in, bf16* __restrict__ dec_x) {
    int i = blockIdx.x * blockDim.x + threadIdx.x;
    if (i < BN_ * D_) dec_x[i] = (bf16)init_in[i % D_];
}
__global__ void init_mask_lps_kernel(const float* __restrict__ mask,
                                     float* __restrict__ mask_dec, float* __restrict__ lps) {
    int i = blockIdx.x * blockDim.x + threadIdx.x;
    if (i >= BN_ * S_) return;
    int bn = i / S_, s = i % S_;
    int b = bn / NS_;
    mask_dec[i] = mask[(size_t)b * S_ + s];
    if (i < BN_) lps[i] = 0.0f;
}

// ---------------- deterministic hash RNG ----------------
__device__ inline uint32_t hash3(uint32_t a, uint32_t b, uint32_t c) {
    uint32_t v = a * 0x9E3779B1u ^ b * 0x85EBCA77u ^ c * 0xC2B2AE3Du;
    v ^= v >> 16; v *= 0x7FEB352Du;
    v ^= v >> 15; v *= 0x846CA68Bu;
    v ^= v >> 16;
    return v;
}

// =====================================================================
// Fused attention + log-softmax + Gumbel sample + state update.
// One block (256 threads = 8 waves) per bn row. enc_base is bf16 [S,B,H]
// (NOT NS-repeated: sample rows share the base-batch encoder row via L2).
//   att[s] = CLIP * tanh( sum_h v[h] * tanh(dec_proj[bn,h] + enc_base[s,b,h]) )
// =====================================================================
__global__ void attn_sample_kernel(int t,
                                   const float* __restrict__ dec_proj,   // [Bn,H]
                                   const bf16*  __restrict__ enc_base,   // [S,B,H]
                                   const float* __restrict__ attn_v,     // [H]
                                   const float* __restrict__ mask,       // [B,S]
                                   float* __restrict__ mask_dec,         // [Bn,S]
                                   float* __restrict__ lps,              // [Bn]
                                   int*   __restrict__ sel_all,          // [S,Bn]
                                   bf16*  __restrict__ dec_x,            // [Bn,D]
                                   const float* __restrict__ input) {    // [S,B,D]
    int bn   = blockIdx.x;
    int b    = bn / NS_;
    int tid  = threadIdx.x;
    int lane = tid & 31;
    int wave = tid >> 5;

    __shared__ float satt[S_];
    __shared__ int   ssel;

    float dp[8], vv[8];
#pragma unroll
    for (int j = 0; j < 8; ++j) {
        int hh = lane + 32 * j;
        dp[j] = dec_proj[(size_t)bn * H_ + hh];
        vv[j] = attn_v[hh];
    }

    // each wave owns one score row s at a time
    for (int si = 0; si < S_ / 8; ++si) {
        int s = si * 8 + wave;
        const bf16* er = enc_base + ((size_t)s * B_ + b) * H_;
        if (si + 1 < S_ / 8)
            __builtin_prefetch(enc_base + ((size_t)(s + 8) * B_ + b) * H_, 0, 1);
        float acc = 0.0f;
#pragma unroll
        for (int j = 0; j < 8; ++j) {
            int hh = lane + 32 * j;
            acc += tanhf(dp[j] + (float)er[hh]) * vv[j];
        }
#pragma unroll
        for (int off = 16; off > 0; off >>= 1) acc += __shfl_xor(acc, off, 32);
        if (lane == 0) satt[s] = CLIP_ * tanhf(acc);
    }
    __syncthreads();

    if (wave == 0) {
        float att[4], key[4];
        float vmax = NEGINF_;
#pragma unroll
        for (int q = 0; q < 4; ++q) {
            int s = lane + 32 * q;
            float md = mask_dec[(size_t)bn * S_ + s];
            float a = (md == 0.0f) ? NEGINF_ : satt[s];
            att[q] = a;
            uint32_t hsh = hash3((uint32_t)t, (uint32_t)bn, (uint32_t)s);
            float u = ((float)hsh + 0.5f) * (1.0f / 4294967296.0f);
            float g = -logf(-logf(u));                 // Gumbel(0,1)
            key[q] = (a <= NEGINF_) ? NEGINF_ : a + g;
            vmax = fmaxf(vmax, a);
        }
#pragma unroll
        for (int off = 16; off > 0; off >>= 1) vmax = fmaxf(vmax, __shfl_xor(vmax, off, 32));
        float sum = 0.0f;
#pragma unroll
        for (int q = 0; q < 4; ++q)
            sum += (att[q] <= NEGINF_) ? 0.0f : __expf(att[q] - vmax);
#pragma unroll
        for (int off = 16; off > 0; off >>= 1) sum += __shfl_xor(sum, off, 32);

        // argmax of perturbed logits
        float bk = NEGINF_; int bs = 0;
#pragma unroll
        for (int q = 0; q < 4; ++q) {
            if (key[q] > bk) { bk = key[q]; bs = lane + 32 * q; }
        }
#pragma unroll
        for (int off = 16; off > 0; off >>= 1) {
            float ok = __shfl_xor(bk, off, 32);
            int   os = __shfl_xor(bs, off, 32);
            if (ok > bk || (ok == bk && os < bs)) { bk = ok; bs = os; }
        }
        if (lane == 0) {
            int sel = bs;
            float logZ = vmax + logf(sum);
            float lp = satt[sel] - logZ;                      // selected is always unmasked
            lps[bn] += lp * mask[(size_t)b * S_ + t];
            mask_dec[(size_t)bn * S_ + sel] = 0.0f;
            sel_all[(size_t)t * BN_ + bn] = sel;
            ssel = sel;
        }
    }
    __syncthreads();

    // gather next decoder input: dec_x[bn,:] = input[sel, b, :]
    int sel = ssel;
    dec_x[(size_t)bn * D_ + tid] = (bf16)input[((size_t)sel * B_ + b) * D_ + tid];
}

// =====================================================================
// finalize: reorder mappings [Bn,S]->(ns,B) rows and lps, write to d_out (float)
// =====================================================================
__global__ void finalize_kernel(const int* __restrict__ sel_all, const float* __restrict__ lps,
                                const float* __restrict__ mask, float* __restrict__ out) {
    int i = blockIdx.x * blockDim.x + threadIdx.x;
    if (i >= BN_ * S_) return;
    int bn = i / S_, s = i % S_;
    int b = bn / NS_, j = bn % NS_;
    int row = j * B_ + b;
    float mr = mask[(size_t)b * S_ + s];
    int sel = sel_all[(size_t)s * BN_ + bn];
    out[(size_t)row * S_ + s] = (mr == 0.0f) ? -1.0f : (float)sel;
    if (s == 0) out[(size_t)BN_ * S_ + row] = lps[bn];
}

// =====================================================================
// host driver
// =====================================================================
extern "C" void kernel_launch(void* const* d_in, const int* in_sizes, int n_in,
                              void* d_out, int out_size, void* d_ws, size_t ws_size,
                              hipStream_t stream) {
    (void)in_sizes; (void)n_in; (void)out_size; (void)ws_size;

    const float* input   = (const float*)d_in[0];   // [S,B,D]
    const float* mask    = (const float*)d_in[1];   // [B,S]
    const float* attn_Wd = (const float*)d_in[2];   // [H,H]
    const float* attn_We = (const float*)d_in[3];   // [H,H]
    const float* attn_b  = (const float*)d_in[4];   // [H]
    const float* attn_v  = (const float*)d_in[5];   // [H]
    const float* enc_Wih = (const float*)d_in[6];   // [L,4H,H]
    const float* enc_Whh = (const float*)d_in[7];
    const float* enc_b   = (const float*)d_in[8];   // [L,4H]
    const float* dec_Wih = (const float*)d_in[9];
    const float* dec_Whh = (const float*)d_in[10];
    const float* dec_b   = (const float*)d_in[11];
    const float* init_di = (const float*)d_in[12];  // [1,D]
    float* out = (float*)d_out;

    // ---- workspace carve (~50 MB) ----
    char* p = (char*)d_ws;
    auto carve = [&](size_t bytes) -> char* {
        char* r = p;
        p += (bytes + 255) & ~(size_t)255;
        return r;
    };
    const size_t Wn  = (size_t)L_ * H4_ * H_;       // 524288 per weight stack
    const size_t Awn = (size_t)H_ * H_;             // 65536
    const size_t Xn  = (size_t)S_ * B_ * D_;        // 4194304
    const size_t En  = (size_t)S_ * B_ * H_;        // 4194304

    bf16* encWih_b = (bf16*)carve(Wn * 2);
    bf16* encWhh_b = (bf16*)carve(Wn * 2);
    bf16* decWih_b = (bf16*)carve(Wn * 2);
    bf16* decWhh_b = (bf16*)carve(Wn * 2);
    bf16* attnWd_b = (bf16*)carve(Awn * 2);
    bf16* attnWe_b = (bf16*)carve(Awn * 2);
    bf16* Xb       = (bf16*)carve(Xn * 2);
    bf16* encout_b = (bf16*)carve(En * 2);
    bf16* encbase_b= (bf16*)carve(En * 2);          // 8.4 MB, L2-resident
    float* z       = (float*)carve((size_t)BN_ * H4_ * 4);
    bf16*  h_enc   = (bf16*)carve((size_t)L_ * B_ * H_ * 2);
    float* c_enc   = (float*)carve((size_t)L_ * B_ * H_ * 4);
    bf16*  h_dec   = (bf16*)carve((size_t)L_ * BN_ * H_ * 2);
    float* c_dec   = (float*)carve((size_t)L_ * BN_ * H_ * 4);
    bf16*  dec_x   = (bf16*)carve((size_t)BN_ * D_ * 2);
    float* dec_proj= (float*)carve((size_t)BN_ * H_ * 4);
    float* mask_dec= (float*)carve((size_t)BN_ * S_ * 4);
    float* lps     = (float*)carve((size_t)BN_ * 4);
    int*   sel_all = (int*)carve((size_t)S_ * BN_ * 4);

    auto ewg = [](size_t n) { return dim3((unsigned)((n + 255) / 256)); };

    // ---- weight / input conversion to bf16 ----
    f32_to_bf16_kernel<<<ewg(Wn), 256, 0, stream>>>(enc_Wih, encWih_b, (int)Wn);
    f32_to_bf16_kernel<<<ewg(Wn), 256, 0, stream>>>(enc_Whh, encWhh_b, (int)Wn);
    f32_to_bf16_kernel<<<ewg(Wn), 256, 0, stream>>>(dec_Wih, decWih_b, (int)Wn);
    f32_to_bf16_kernel<<<ewg(Wn), 256, 0, stream>>>(dec_Whh, decWhh_b, (int)Wn);
    f32_to_bf16_kernel<<<ewg(Awn), 256, 0, stream>>>(attn_Wd, attnWd_b, (int)Awn);
    f32_to_bf16_kernel<<<ewg(Awn), 256, 0, stream>>>(attn_We, attnWe_b, (int)Awn);
    f32_to_bf16_kernel<<<ewg(Xn), 256, 0, stream>>>(input, Xb, (int)Xn);

    // ---- encoder ----
    zero_bf16_kernel<<<ewg((size_t)L_ * B_ * H_), 256, 0, stream>>>(h_enc, L_ * B_ * H_);
    zero_f32_kernel<<<ewg((size_t)L_ * B_ * H_), 256, 0, stream>>>(c_enc, L_ * B_ * H_);

    // per-step encoder GEMM: small M -> 16x16 tiles for max wave parallelism
    dim3 genc(H4_ / 64, B_ / 16);                   // (16, 8)
    for (int t = 0; t < S_; ++t) {
        for (int l = 0; l < L_; ++l) {
            const bf16* A1 = (l == 0) ? (Xb + (size_t)t * B_ * D_) : h_enc;
            gemm_wmma<1, true, false><<<genc, 128, 0, stream>>>(
                A1, encWih_b + (size_t)l * H4_ * H_,
                h_enc + (size_t)l * B_ * H_, encWhh_b + (size_t)l * H4_ * H_,
                enc_b + (size_t)l * H4_,
                z, B_, H4_, H_);
            lstm_cell_kernel<<<ewg((size_t)B_ * H_), 256, 0, stream>>>(
                z, c_enc + (size_t)l * B_ * H_, h_enc + (size_t)l * B_ * H_,
                (l == L_ - 1) ? (encout_b + (size_t)t * B_ * H_) : nullptr,
                B_ * H_);
        }
    }

    // enc_base[s,b,:] = enc_out[s,b,:] @ We^T + attn_b   (bf16 output, L2-resident)
    gemm_wmma<4, false, true><<<dim3(H_ / 256, (S_ * B_) / 16), 128, 0, stream>>>(
        encout_b, attnWe_b, nullptr, nullptr, attn_b,
        encbase_b, S_ * B_, H_, H_);

    // ---- decoder init ----
    init_dec_state_kernel<<<ewg((size_t)L_ * BN_ * H_), 256, 0, stream>>>(h_enc, c_enc, h_dec, c_dec);
    init_decx_kernel<<<ewg((size_t)BN_ * D_), 256, 0, stream>>>(init_di, dec_x);
    init_mask_lps_kernel<<<ewg((size_t)BN_ * S_), 256, 0, stream>>>(mask, mask_dec, lps);

    // decoder GEMMs: M=2048 -> 16x64 tiles per wave (A-frag reused 4x)
    dim3 gdec(H4_ / 256, BN_ / 16);                 // (4, 128)
    dim3 gprj(H_ / 256, BN_ / 16);                  // (1, 128)
    for (int t = 0; t < S_; ++t) {
        // layer 0: z = dec_x@Wih0^T + h0@Whh0^T + b0
        gemm_wmma<4, true, false><<<gdec, 128, 0, stream>>>(
            dec_x, decWih_b, h_dec, decWhh_b, dec_b,
            z, BN_, H4_, H_);
        lstm_cell_kernel<<<ewg((size_t)BN_ * H_), 256, 0, stream>>>(
            z, c_dec, h_dec, nullptr, BN_ * H_);
        // layer 1: z = h0@Wih1^T + h1@Whh1^T + b1
        gemm_wmma<4, true, false><<<gdec, 128, 0, stream>>>(
            h_dec, decWih_b + (size_t)H4_ * H_,
            h_dec + (size_t)BN_ * H_, decWhh_b + (size_t)H4_ * H_,
            dec_b + H4_,
            z, BN_, H4_, H_);
        lstm_cell_kernel<<<ewg((size_t)BN_ * H_), 256, 0, stream>>>(
            z, c_dec + (size_t)BN_ * H_, h_dec + (size_t)BN_ * H_, nullptr, BN_ * H_);
        // dec_proj = h1 @ Wd^T
        gemm_wmma<4, false, false><<<gprj, 128, 0, stream>>>(
            h_dec + (size_t)BN_ * H_, attnWd_b, nullptr, nullptr, nullptr,
            dec_proj, BN_, H_, H_);
        // fused attention + sample + state update
        attn_sample_kernel<<<BN_, 256, 0, stream>>>(
            t, dec_proj, encbase_b, attn_v, mask, mask_dec, lps, sel_all, dec_x, input);
    }

    finalize_kernel<<<ewg((size_t)BN_ * S_), 256, 0, stream>>>(sel_all, lps, mask, out);
}
